// SimpleTransformerBlock_69939247448556
// MI455X (gfx1250) — matrix operand order repaired and checked by reading, not compiled, not run on hardware
//
#include <hip/hip_runtime.h>

// ---------- types ----------
typedef __attribute__((ext_vector_type(16))) __bf16   v16bf;
typedef __attribute__((ext_vector_type(8)))  float    v8f;
typedef __attribute__((ext_vector_type(4)))  unsigned int v4u;
typedef __attribute__((ext_vector_type(4)))  int      v4i;

typedef unsigned short u16;
typedef long long ll;

typedef __attribute__((address_space(1))) v4i gv4i;   // global <4 x i32>
typedef __attribute__((address_space(3))) v4i sv4i;   // LDS    <4 x i32>

// ---------- constants ----------
#define BATCH 4
#define SEQ   2048
#define DMODEL 1024
#define DFF   4096
#define MTOT  (BATCH * SEQ)   // 8192

#define BM 128
#define BN 256
#define BK 32
#define LDST 40               // padded LDS row stride (u16); 80B keeps 16B alignment

// ---------- async global->LDS support (guarded: falls back to sync path) ----------
#if defined(__HIP_DEVICE_COMPILE__) && __has_builtin(__builtin_amdgcn_global_load_async_to_lds_b128)
#define USE_ASYNC 1
#else
#define USE_ASYNC 0
#endif

template <int OFS>
__device__ __forceinline__ void async_cp_b128(const u16* g, u16* l) {
#if USE_ASYNC
    const v4i* gg = reinterpret_cast<const v4i*>(g);
    v4i*       lg = reinterpret_cast<v4i*>(l);
    __builtin_amdgcn_global_load_async_to_lds_b128((gv4i*)gg, (sv4i*)lg, OFS, 0);
#else
    (void)g; (void)l;
#endif
}

template <int N>
__device__ __forceinline__ void wait_asynccnt() {
#if USE_ASYNC
#if __has_builtin(__builtin_amdgcn_s_wait_asynccnt)
    __builtin_amdgcn_s_wait_asynccnt(N);
#else
    asm volatile("s_wait_asynccnt %0" ::"n"(N) : "memory");
#endif
#endif
}

// ---------- helpers ----------
__device__ __forceinline__ u16 f2bf(float f) {
    unsigned int u = __builtin_bit_cast(unsigned int, f);
    u = u + 0x7FFFu + ((u >> 16) & 1u);   // round-to-nearest-even
    return (u16)(u >> 16);
}
__device__ __forceinline__ float gelu_exact(float x) {
    return 0.5f * x * (1.0f + erff(x * 0.70710678118654752f));
}

union FragU { v4u u[2]; v16bf v; };

// A fragment: 16x32 bf16 (ISA 7.12.2): lane<16 K{0..7,16..23}, lane>=16 K{8..15,24..31}
__device__ __forceinline__ v16bf load_fragA(const u16* S, int row0, int lane) {
    const u16* p = S + (row0 + (lane & 15)) * LDST + ((lane & 16) ? 8 : 0);
    FragU r;
    r.u[0] = *reinterpret_cast<const v4u*>(p);
    r.u[1] = *reinterpret_cast<const v4u*>(p + 16);
    return r.v;
}
// B fragment: 32x16 bf16: lane<16 K{0..7,16..23}, lane>=16 K{8..15,24..31} (S holds B^T [N][K])
__device__ __forceinline__ v16bf load_fragB(const u16* S, int col0, int lane) {
    const u16* p = S + (col0 + (lane & 15)) * LDST + ((lane & 16) ? 8 : 0);
    FragU r;
    r.u[0] = *reinterpret_cast<const v4u*>(p);
    r.u[1] = *reinterpret_cast<const v4u*>(p + 16);
    return r.v;
}

// ---------- epilogue variants ----------
#define EPI_PLAIN_BF16     0
#define EPI_BIAS_BF16      1
#define EPI_SCALE_F32      2
#define EPI_BIAS_RES_F32   3
#define EPI_BIAS_GELU_BF16 4

// C[M,N] = A[M,K] @ Bt[N,K]^T ; A,Bt bf16 row-major. Batched via blockIdx.z.
// 256 threads = 8 waves (2x4); each wave computes a 64x64 patch = 16 WMMA/K-step.
template <int EPI>
__global__ __launch_bounds__(256)
void gemm_wmma_bf16(const u16* __restrict__ A, ll strideA,
                    const u16* __restrict__ Bt, ll strideB,
                    void* __restrict__ Cout, ll strideC,
                    const float* __restrict__ bias,
                    const float* __restrict__ residual,
                    float scale, int M, int N, int K) {
    constexpr bool HAS_BIAS = (EPI == EPI_BIAS_BF16 || EPI == EPI_BIAS_RES_F32 || EPI == EPI_BIAS_GELU_BF16);
    constexpr bool HAS_RES  = (EPI == EPI_BIAS_RES_F32);
    constexpr bool DO_GELU  = (EPI == EPI_BIAS_GELU_BF16);
    constexpr bool DO_SCALE = (EPI == EPI_SCALE_F32);
    constexpr bool F32OUT   = (EPI == EPI_SCALE_F32 || EPI == EPI_BIAS_RES_F32);

#if USE_ASYNC
    __shared__ u16 As[2 * BM * LDST];
    __shared__ u16 Bs[2 * BN * LDST];
#else
    __shared__ u16 As[BM * LDST];
    __shared__ u16 Bs[BN * LDST];
#endif

    const int b = blockIdx.z;
    A  += (ll)b * strideA;
    Bt += (ll)b * strideB;

    const int tid  = threadIdx.x;
    const int lane = tid & 31;
    const int wid  = tid >> 5;            // 0..7
    const int wm   = (wid >> 2) * 64;     // wave row base in tile (0/64)
    const int wn   = (wid & 3) * 64;      // wave col base in tile (0..192)
    const int m0   = blockIdx.x * BM;
    const int n0   = blockIdx.y * BN;

    // per-thread load assignment: 32B chunks
    const int ra  = tid >> 1;             // A row 0..127
    const int rb0 = tid >> 1;             // B rows (two per thread)
    const int rb1 = rb0 + 128;
    const int kh  = (tid & 1) * 16;       // k sub-chunk (elems)

    v8f acc[4][4] = {};

    auto compute_stage = [&](const u16* Asp, const u16* Bsp) {
        v16bf af[4], bfm[4];
#pragma unroll
        for (int i = 0; i < 4; ++i) af[i]  = load_fragA(Asp, wm + i * 16, lane);
#pragma unroll
        for (int j = 0; j < 4; ++j) bfm[j] = load_fragB(Bsp, wn + j * 16, lane);
#pragma unroll
        for (int i = 0; i < 4; ++i)
#pragma unroll
            for (int j = 0; j < 4; ++j)
                acc[i][j] = __builtin_amdgcn_wmma_f32_16x16x32_bf16(
                    false, af[i], false, bfm[j], (short)0, acc[i][j], false, false);
    };

#if USE_ASYNC
    auto issue_tile = [&](int kt, int st) {
        const int k0 = kt * BK;
        u16* Asp = &As[st * (BM * LDST)];
        u16* Bsp = &Bs[st * (BN * LDST)];
        const u16* gA  = A  + (ll)(m0 + ra)  * K + k0 + kh;
        const u16* gB0 = Bt + (ll)(n0 + rb0) * K + k0 + kh;
        const u16* gB1 = Bt + (ll)(n0 + rb1) * K + k0 + kh;
        async_cp_b128<0>(gA, Asp + ra * LDST + kh);
        async_cp_b128<16>(gA, Asp + ra * LDST + kh);
        async_cp_b128<0>(gB0, Bsp + rb0 * LDST + kh);
        async_cp_b128<16>(gB0, Bsp + rb0 * LDST + kh);
        async_cp_b128<0>(gB1, Bsp + rb1 * LDST + kh);
        async_cp_b128<16>(gB1, Bsp + rb1 * LDST + kh);
    };

    const int NT = K / BK;
    issue_tile(0, 0);                         // prologue: tile 0 -> stage 0
    for (int kt = 0; kt < NT; ++kt) {
        const int st = kt & 1;
        __syncthreads();                      // all waves done reading stage st^1
        if (kt + 1 < NT) {
            issue_tile(kt + 1, st ^ 1);       // stream next tile while computing
            wait_asynccnt<6>();               // 6 newer pending -> tile kt landed
        } else {
            wait_asynccnt<0>();
        }
        __syncthreads();                      // tile kt visible to all waves
        compute_stage(&As[st * (BM * LDST)], &Bs[st * (BN * LDST)]);
    }
#else
    for (int k0 = 0; k0 < K; k0 += BK) {
        const v4u* ga  = reinterpret_cast<const v4u*>(A  + (ll)(m0 + ra)  * K + k0 + kh);
        const v4u* gb0 = reinterpret_cast<const v4u*>(Bt + (ll)(n0 + rb0) * K + k0 + kh);
        const v4u* gb1 = reinterpret_cast<const v4u*>(Bt + (ll)(n0 + rb1) * K + k0 + kh);
        v4u a0 = ga[0],  a1 = ga[1];
        v4u b0 = gb0[0], b1 = gb0[1];
        v4u b2 = gb1[0], b3 = gb1[1];
        if (k0 + BK < K) {
            __builtin_prefetch(reinterpret_cast<const char*>(ga)  + BK * 2, 0, 1);
            __builtin_prefetch(reinterpret_cast<const char*>(gb0) + BK * 2, 0, 1);
            __builtin_prefetch(reinterpret_cast<const char*>(gb1) + BK * 2, 0, 1);
        }
        __syncthreads();
        *reinterpret_cast<v4u*>(&As[ra  * LDST + kh])     = a0;
        *reinterpret_cast<v4u*>(&As[ra  * LDST + kh + 8]) = a1;
        *reinterpret_cast<v4u*>(&Bs[rb0 * LDST + kh])     = b0;
        *reinterpret_cast<v4u*>(&Bs[rb0 * LDST + kh + 8]) = b1;
        *reinterpret_cast<v4u*>(&Bs[rb1 * LDST + kh])     = b2;
        *reinterpret_cast<v4u*>(&Bs[rb1 * LDST + kh + 8]) = b3;
        __syncthreads();
        compute_stage(As, Bs);
    }
#endif

    // ---- epilogue: C layout: VGPR r -> M = r + (lane>=16 ? 8 : 0), N = lane&15
    const int colb = n0 + wn + (lane & 15);
    const int rowb = m0 + wm + ((lane & 16) ? 8 : 0);
#pragma unroll
    for (int i = 0; i < 4; ++i) {
#pragma unroll
        for (int j = 0; j < 4; ++j) {
            const int c = colb + j * 16;
            float bia = 0.0f;
            if (HAS_BIAS) bia = bias[c];
#pragma unroll
            for (int r = 0; r < 8; ++r) {
                const int m = rowb + i * 16 + r;
                float v = acc[i][j][r];
                if (DO_SCALE) v *= scale;
                if (HAS_BIAS) v += bia;
                if (DO_GELU)  v = gelu_exact(v);
                const ll idx = (ll)b * strideC + (ll)m * N + c;
                if (HAS_RES) v += residual[(ll)m * N + c];
                if (F32OUT) reinterpret_cast<float*>(Cout)[idx] = v;
                else        reinterpret_cast<u16*>(Cout)[idx]   = f2bf(v);
            }
        }
    }
}

// ---------- LayerNorm (fp32 in -> bf16 out), one block per row of D=1024 ----------
__global__ __launch_bounds__(256)
void layernorm_bf16(const float* __restrict__ x, const float* __restrict__ g,
                    const float* __restrict__ be, u16* __restrict__ out) {
    const int row = blockIdx.x;
    const float* xr = x + (ll)row * DMODEL;
    float vals[4];
    float s = 0.f, ss = 0.f;
#pragma unroll
    for (int i = 0; i < 4; ++i) {
        float v = xr[threadIdx.x + i * 256];
        vals[i] = v; s += v; ss += v * v;
    }
#pragma unroll
    for (int off = 16; off > 0; off >>= 1) {
        s  += __shfl_down(s,  off, 32);
        ss += __shfl_down(ss, off, 32);
    }
    __shared__ float red[2][8];
    const int lane = threadIdx.x & 31, wid = threadIdx.x >> 5;
    if (lane == 0) { red[0][wid] = s; red[1][wid] = ss; }
    __syncthreads();
    s = 0.f; ss = 0.f;
#pragma unroll
    for (int i = 0; i < 8; ++i) { s += red[0][i]; ss += red[1][i]; }
    const float mu  = s * (1.0f / DMODEL);
    const float var = ss * (1.0f / DMODEL) - mu * mu;
    const float inv = rsqrtf(var + 1e-5f);
#pragma unroll
    for (int i = 0; i < 4; ++i) {
        const int c = threadIdx.x + i * 256;
        out[(ll)row * DMODEL + c] = f2bf((vals[i] - mu) * inv * g[c] + be[c]);
    }
}

// ---------- row softmax (fp32 [rows][SEQ] -> bf16) ----------
__global__ __launch_bounds__(256)
void softmax_bf16(const float* __restrict__ sc, u16* __restrict__ out) {
    const int row = blockIdx.x;
    const float* sr = sc + (ll)row * SEQ;
    float v[8];
    float mx = -3.4e38f;
#pragma unroll
    for (int i = 0; i < 8; ++i) { v[i] = sr[threadIdx.x + i * 256]; mx = fmaxf(mx, v[i]); }
#pragma unroll
    for (int off = 16; off > 0; off >>= 1) mx = fmaxf(mx, __shfl_down(mx, off, 32));
    __shared__ float red[8];
    const int lane = threadIdx.x & 31, wid = threadIdx.x >> 5;
    if (lane == 0) red[wid] = mx;
    __syncthreads();
    mx = red[0];
#pragma unroll
    for (int i = 1; i < 8; ++i) mx = fmaxf(mx, red[i]);
    __syncthreads();
    float s = 0.f;
#pragma unroll
    for (int i = 0; i < 8; ++i) { v[i] = __expf(v[i] - mx); s += v[i]; }
#pragma unroll
    for (int off = 16; off > 0; off >>= 1) s += __shfl_down(s, off, 32);
    if (lane == 0) red[wid] = s;
    __syncthreads();
    s = 0.f;
#pragma unroll
    for (int i = 0; i < 8; ++i) s += red[i];
    const float inv = 1.0f / s;
#pragma unroll
    for (int i = 0; i < 8; ++i)
        out[(ll)row * SEQ + threadIdx.x + i * 256] = f2bf(v[i] * inv);
}

// ---------- transpose fp32 [R,C] -> bf16 [C,R] (weights) ----------
__global__ __launch_bounds__(256)
void transpose_f32_to_bf16T(const float* __restrict__ src, u16* __restrict__ dst, int R, int C) {
    __shared__ float tile[32][33];
    const int c0 = blockIdx.x * 32, r0 = blockIdx.y * 32;
    const int tx = threadIdx.x & 31, ty = threadIdx.x >> 5;  // 32 x 8
#pragma unroll
    for (int i = 0; i < 32; i += 8)
        tile[ty + i][tx] = src[(ll)(r0 + ty + i) * C + c0 + tx];
    __syncthreads();
#pragma unroll
    for (int i = 0; i < 32; i += 8)
        dst[(ll)(c0 + ty + i) * R + r0 + tx] = f2bf(tile[tx][ty + i]);
}

// ---------- transpose bf16 [R,C] -> bf16 [C,R], batched ----------
__global__ __launch_bounds__(256)
void transpose_bf16T(const u16* __restrict__ src, u16* __restrict__ dst, int R, int C) {
    __shared__ u16 tile[32][34];
    const ll bs = (ll)blockIdx.z * R * C;
    const int c0 = blockIdx.x * 32, r0 = blockIdx.y * 32;
    const int tx = threadIdx.x & 31, ty = threadIdx.x >> 5;
#pragma unroll
    for (int i = 0; i < 32; i += 8)
        tile[ty + i][tx] = src[bs + (ll)(r0 + ty + i) * C + c0 + tx];
    __syncthreads();
#pragma unroll
    for (int i = 0; i < 32; i += 8)
        dst[bs + (ll)(c0 + ty + i) * R + r0 + tx] = tile[tx][ty + i];
}

// ---------- host side ----------
extern "C" void kernel_launch(void* const* d_in, const int* in_sizes, int n_in,
                              void* d_out, int out_size, void* d_ws, size_t ws_size,
                              hipStream_t stream) {
    (void)in_sizes; (void)n_in; (void)out_size; (void)ws_size;

    const float* x   = (const float*)d_in[0];
    const float* wq  = (const float*)d_in[1];
    const float* bq  = (const float*)d_in[2];
    const float* wk  = (const float*)d_in[3];
    const float* bk  = (const float*)d_in[4];
    const float* wv  = (const float*)d_in[5];
    const float* bv  = (const float*)d_in[6];
    const float* wo  = (const float*)d_in[7];
    const float* bo  = (const float*)d_in[8];
    const float* w1  = (const float*)d_in[9];
    const float* b1  = (const float*)d_in[10];
    const float* w2  = (const float*)d_in[11];
    const float* b2  = (const float*)d_in[12];
    const float* g1  = (const float*)d_in[13];
    const float* be1 = (const float*)d_in[14];
    const float* g2  = (const float*)d_in[15];
    const float* be2 = (const float*)d_in[16];
    float* out = (float*)d_out;

    char* w = (char*)d_ws;
    auto carve = [&](size_t bytes) { char* p = w; w += (bytes + 255) & ~(size_t)255; return p; };
    u16*   WqT   = (u16*)  carve((size_t)DMODEL * DMODEL * 2);
    u16*   WkT   = (u16*)  carve((size_t)DMODEL * DMODEL * 2);
    u16*   WvT   = (u16*)  carve((size_t)DMODEL * DMODEL * 2);
    u16*   WoT   = (u16*)  carve((size_t)DMODEL * DMODEL * 2);
    u16*   W1T   = (u16*)  carve((size_t)DFF * DMODEL * 2);
    u16*   W2T   = (u16*)  carve((size_t)DMODEL * DFF * 2);
    u16*   h1    = (u16*)  carve((size_t)MTOT * DMODEL * 2);
    u16*   Qb    = (u16*)  carve((size_t)MTOT * DMODEL * 2);
    u16*   Kb    = (u16*)  carve((size_t)MTOT * DMODEL * 2);
    u16*   Vb    = (u16*)  carve((size_t)MTOT * DMODEL * 2);
    u16*   Vt    = (u16*)  carve((size_t)BATCH * DMODEL * SEQ * 2);
    float* scores= (float*)carve((size_t)BATCH * SEQ * SEQ * 4);
    u16*   attn  = (u16*)  carve((size_t)BATCH * SEQ * SEQ * 2);
    u16*   ctx   = (u16*)  carve((size_t)MTOT * DMODEL * 2);
    float* x2    = (float*)carve((size_t)MTOT * DMODEL * 4);
    u16*   h2    = (u16*)  carve((size_t)MTOT * DMODEL * 2);
    u16*   mbuf  = (u16*)  carve((size_t)MTOT * DFF * 2);

    const dim3 blk(256);

    // 1) weights -> bf16 transposed [N,K]
    transpose_f32_to_bf16T<<<dim3(DMODEL/32, DMODEL/32), blk, 0, stream>>>(wq, WqT, DMODEL, DMODEL);
    transpose_f32_to_bf16T<<<dim3(DMODEL/32, DMODEL/32), blk, 0, stream>>>(wk, WkT, DMODEL, DMODEL);
    transpose_f32_to_bf16T<<<dim3(DMODEL/32, DMODEL/32), blk, 0, stream>>>(wv, WvT, DMODEL, DMODEL);
    transpose_f32_to_bf16T<<<dim3(DMODEL/32, DMODEL/32), blk, 0, stream>>>(wo, WoT, DMODEL, DMODEL);
    transpose_f32_to_bf16T<<<dim3(DFF/32,    DMODEL/32), blk, 0, stream>>>(w1, W1T, DMODEL, DFF);
    transpose_f32_to_bf16T<<<dim3(DMODEL/32, DFF/32),    blk, 0, stream>>>(w2, W2T, DFF, DMODEL);

    // 2) LN1
    layernorm_bf16<<<dim3(MTOT), blk, 0, stream>>>(x, g1, be1, h1);

    // 3) Q,K,V projections
    const dim3 gproj(MTOT/BM, DMODEL/BN, 1);
    gemm_wmma_bf16<EPI_BIAS_BF16><<<gproj, blk, 0, stream>>>(h1, 0, WqT, 0, Qb, 0, bq, nullptr, 1.f, MTOT, DMODEL, DMODEL);
    gemm_wmma_bf16<EPI_BIAS_BF16><<<gproj, blk, 0, stream>>>(h1, 0, WkT, 0, Kb, 0, bk, nullptr, 1.f, MTOT, DMODEL, DMODEL);
    gemm_wmma_bf16<EPI_BIAS_BF16><<<gproj, blk, 0, stream>>>(h1, 0, WvT, 0, Vb, 0, bv, nullptr, 1.f, MTOT, DMODEL, DMODEL);

    // 4) V^T per batch: [SEQ,DMODEL] -> [DMODEL,SEQ]
    transpose_bf16T<<<dim3(DMODEL/32, SEQ/32, BATCH), blk, 0, stream>>>(Vb, Vt, SEQ, DMODEL);

    // 5) scores = Q K^T * (1/sqrt(64)) ; Bt = K (already [t,d] row-major)
    gemm_wmma_bf16<EPI_SCALE_F32><<<dim3(SEQ/BM, SEQ/BN, BATCH), blk, 0, stream>>>(
        Qb, (ll)SEQ * DMODEL, Kb, (ll)SEQ * DMODEL, scores, (ll)SEQ * SEQ,
        nullptr, nullptr, 0.125f, SEQ, SEQ, DMODEL);

    // 6) softmax rows
    softmax_bf16<<<dim3(BATCH * SEQ), blk, 0, stream>>>(scores, attn);

    // 7) ctx = attn @ V ; Bt = V^T
    gemm_wmma_bf16<EPI_PLAIN_BF16><<<dim3(SEQ/BM, DMODEL/BN, BATCH), blk, 0, stream>>>(
        attn, (ll)SEQ * SEQ, Vt, (ll)DMODEL * SEQ, ctx, (ll)SEQ * DMODEL,
        nullptr, nullptr, 1.f, SEQ, DMODEL, SEQ);

    // 8) x2 = ctx @ Wo + bo + x
    gemm_wmma_bf16<EPI_BIAS_RES_F32><<<gproj, blk, 0, stream>>>(
        ctx, 0, WoT, 0, x2, 0, bo, x, 1.f, MTOT, DMODEL, DMODEL);

    // 9) LN2
    layernorm_bf16<<<dim3(MTOT), blk, 0, stream>>>(x2, g2, be2, h2);

    // 10) m = gelu(h2 @ W1 + b1)
    gemm_wmma_bf16<EPI_BIAS_GELU_BF16><<<dim3(MTOT/BM, DFF/BN, 1), blk, 0, stream>>>(
        h2, 0, W1T, 0, mbuf, 0, b1, nullptr, 1.f, MTOT, DFF, DMODEL);

    // 11) out = m @ W2 + b2 + x2
    gemm_wmma_bf16<EPI_BIAS_RES_F32><<<dim3(MTOT/BM, DMODEL/BN, 1), blk, 0, stream>>>(
        mbuf, 0, W2T, 0, out, 0, b2, x2, 1.f, MTOT, DMODEL, DFF);
}
